// QueryBiasedOuterProduct_80178449482192
// MI455X (gfx1250) — matrix operand order repaired and checked by reading, not compiled, not run on hardware
//
#include <hip/hip_runtime.h>
#include <math.h>

// ---- Problem dims (fixed by reference) ----
#define SS   256
#define NNq  256
#define DDm  256
#define CCc  32
#define DPp  128
#define EPSF 1e-8f
#define LAMBDA_INIT 0.2f

typedef __attribute__((ext_vector_type(16))) __bf16 v16bf;
typedef __attribute__((ext_vector_type(8)))  float  v8f;

#if __has_builtin(__builtin_amdgcn_global_load_async_to_lds_b128)
#define HAVE_ASYNC_LDS 1
// Builtin expects: (v4i addrspace(1)*, v4i addrspace(3)*, imm offset, imm cpol)
typedef __attribute__((__vector_size__(16))) int vi4;
typedef __attribute__((address_space(1))) vi4* as1_vi4_p;
typedef __attribute__((address_space(3))) vi4* as3_vi4_p;
#if __has_builtin(__builtin_amdgcn_s_wait_asynccnt)
#define WAIT_ASYNC_4() __builtin_amdgcn_s_wait_asynccnt(4)
#define WAIT_ASYNC_0() __builtin_amdgcn_s_wait_asynccnt(0)
#else
#define WAIT_ASYNC_4() asm volatile("s_wait_asynccnt 0x4" ::: "memory")
#define WAIT_ASYNC_0() asm volatile("s_wait_asynccnt 0x0" ::: "memory")
#endif
#else
#define HAVE_ASYNC_LDS 0
#define WAIT_ASYNC_4()
#define WAIT_ASYNC_0()
#endif

__device__ __forceinline__ float wave_sum32(float v) {
#pragma unroll
  for (int m = 16; m >= 1; m >>= 1) v += __shfl_xor(v, m, 32);
  return v;
}

// =====================================================================
// Kernel 1: LayerNorm of msa_repr (f32 in) -> x_bf16 (for WMMA GEMMs)
//           and x0 row block (s==0) kept in f32 for the q projection.
// One wave (32 lanes) per 256-wide row; 8 rows per block.
// =====================================================================
__global__ void __launch_bounds__(256) ln_kernel(
    const float* __restrict__ msa, const float* __restrict__ lnw,
    const float* __restrict__ lnb, __bf16* __restrict__ xbf,
    float* __restrict__ x0f) {
  int wave = threadIdx.x >> 5, lane = threadIdx.x & 31;
  int row = blockIdx.x * 8 + wave;  // row = s*N + n
  const float* xr = msa + (size_t)row * DDm;
  float4 v0 = *(const float4*)(xr + lane * 8);
  float4 v1 = *(const float4*)(xr + lane * 8 + 4);
  float vals[8] = {v0.x, v0.y, v0.z, v0.w, v1.x, v1.y, v1.z, v1.w};
  float s = 0.f, sq = 0.f;
#pragma unroll
  for (int u = 0; u < 8; ++u) { s += vals[u]; sq += vals[u] * vals[u]; }
  s  = wave_sum32(s);
  sq = wave_sum32(sq);
  float mu  = s * (1.0f / DDm);
  float var = sq * (1.0f / DDm) - mu * mu;
  float r   = rsqrtf(var + 1e-5f);
#pragma unroll
  for (int u = 0; u < 8; ++u) {
    int col = lane * 8 + u;
    float y = (vals[u] - mu) * r * lnw[col] + lnb[col];
    xbf[(size_t)row * DDm + col] = (__bf16)y;
    if (row < NNq) x0f[row * DDm + col] = y;  // s == 0 slice
  }
}

// =====================================================================
// Kernel 2: weight prep -> transposed bf16 copies of k_w and wp
//   kwt[d][o]  = k_w[o][d]   (256 x 256)
//   wpt[k][p]  = wp[p][k]    (1024 x 256)
// =====================================================================
__global__ void wprep_kernel(const float* __restrict__ kw,
                             const float* __restrict__ wp,
                             __bf16* __restrict__ kwt,
                             __bf16* __restrict__ wpt) {
  int t = blockIdx.x * 256 + threadIdx.x;
  if (t < 256 * 256) {
    int d = t >> 8, o = t & 255;
    kwt[t] = (__bf16)kw[o * 256 + d];
  }
  if (t < 1024 * 256) {
    int k = t >> 8, p = t & 255;
    wpt[t] = (__bf16)wp[p * 1024 + k];
  }
}

// =====================================================================
// Tile staging: one 128x32 A chunk + 32x128 B chunk into an LDS buffer.
// Async path uses GLOBAL_LOAD_ASYNC_TO_LDS_B128 (ASYNCcnt, no VGPR
// round-trip); fallback is global_load_b128 + ds_store_b128.
// Each of the 256 threads moves 2x16B of A and 2x16B of B.
// =====================================================================
__device__ __forceinline__ void tile_issue(
    const __bf16* __restrict__ Ag, long lda,
    const __bf16* __restrict__ Bg, long ldb,
    long Mbase, long Nbase, int kk, char* smem, int buf) {
  int t = threadIdx.x;
  int arow = t >> 1, apos = (t & 1) * 16;  // 128 rows x 32 k
  int brow = t >> 3, bpos = (t & 7) * 16;  // 32 k x 128 cols
  __bf16* As = (__bf16*)(smem + buf * 16384);
  __bf16* Bs = (__bf16*)(smem + buf * 16384 + 8192);
  const __bf16* ga = Ag + (Mbase + arow) * lda + kk * 32 + apos;
  const __bf16* gb = Bg + (long)(kk * 32 + brow) * ldb + Nbase + bpos;
#if HAVE_ASYNC_LDS
  __builtin_amdgcn_global_load_async_to_lds_b128(
      (as1_vi4_p)ga, (as3_vi4_p)(As + arow * 32 + apos), 0, 0);
  __builtin_amdgcn_global_load_async_to_lds_b128(
      (as1_vi4_p)(ga + 8), (as3_vi4_p)(As + arow * 32 + apos + 8), 0, 0);
  __builtin_amdgcn_global_load_async_to_lds_b128(
      (as1_vi4_p)gb, (as3_vi4_p)(Bs + brow * 128 + bpos), 0, 0);
  __builtin_amdgcn_global_load_async_to_lds_b128(
      (as1_vi4_p)(gb + 8), (as3_vi4_p)(Bs + brow * 128 + bpos + 8), 0, 0);
#else
  uint4 a0 = ((const uint4*)ga)[0], a1 = ((const uint4*)ga)[1];
  uint4 b0 = ((const uint4*)gb)[0], b1 = ((const uint4*)gb)[1];
  ((uint4*)(As + arow * 32 + apos))[0] = a0;
  ((uint4*)(As + arow * 32 + apos))[1] = a1;
  ((uint4*)(Bs + brow * 128 + bpos))[0] = b0;
  ((uint4*)(Bs + brow * 128 + bpos))[1] = b1;
#endif
}

// =====================================================================
// Double-buffered 128x128xK bf16 WMMA tile GEMM (K in steps of 32).
// 256 threads = 8 waves; wave w owns M-tile w (16 rows) x 8 N-tiles.
// A fragment layout (16-bit A 16x32, ISA 7.12.2):
//   lane<16: M=lane, K in {0..7,16..23}; lane>=16: M=lane-16, K in {8..15,24..31}
// B fragment (32x16): lane holds column N=lane%16, 16 K values in A's order.
// smem must provide 32KB (2 x (8KB A + 8KB B)).
// =====================================================================
__device__ __forceinline__ void gemm128_db(
    const __bf16* __restrict__ Ag, long lda,
    const __bf16* __restrict__ Bg, long ldb,
    long Mbase, long Nbase, int ksteps,
    char* smem, v8f acc[8]) {
  int lane = threadIdx.x & 31, wave = threadIdx.x >> 5;
  int m = lane & 15, half = lane >> 4;
  tile_issue(Ag, lda, Bg, ldb, Mbase, Nbase, 0, smem, 0);
  for (int kk = 0; kk < ksteps; ++kk) {
    int buf = kk & 1;
    if (kk + 1 < ksteps) {
      tile_issue(Ag, lda, Bg, ldb, Mbase, Nbase, kk + 1, smem, buf ^ 1);
      WAIT_ASYNC_4();  // own tile-(kk) ops complete; tile-(kk+1) may be in flight
    } else {
      WAIT_ASYNC_0();
    }
    __syncthreads();   // all waves' tile-(kk) data visible in LDS
    __bf16* As = (__bf16*)(smem + buf * 16384);
    __bf16* Bs = (__bf16*)(smem + buf * 16384 + 8192);
    v16bf af;
#pragma unroll
    for (int e = 0; e < 16; ++e) {
      int k = (e < 8 ? e : e + 8) + half * 8;
      af[e] = As[(wave * 16 + m) * 32 + k];
    }
#pragma unroll
    for (int nt = 0; nt < 8; ++nt) {
      v16bf bf;
#pragma unroll
      for (int e = 0; e < 16; ++e) {
        int k = (e < 8 ? e : e + 8) + half * 8;
        bf[e] = Bs[k * 128 + nt * 16 + m];
      }
      acc[nt] = __builtin_amdgcn_wmma_f32_16x16x32_bf16(
          false, af, false, bf, (short)0, acc[nt], false, false);
    }
    __syncthreads();   // reads done before this buffer is overwritten
  }
}

// =====================================================================
// Kernel 3: k projection GEMM: k_raw = x_bf16 (65536x256) @ kwt (256x256)
// =====================================================================
__global__ void __launch_bounds__(256) kproj_gemm(
    const __bf16* __restrict__ xbf, const __bf16* __restrict__ kwt,
    float* __restrict__ kraw) {
  __shared__ __align__(16) char smem[32768];
  v8f acc[8];
#pragma unroll
  for (int i = 0; i < 8; ++i) { v8f z = {}; acc[i] = z; }
  long Mbase = (long)blockIdx.x * 128, Nbase = (long)blockIdx.y * 128;
  gemm128_db(xbf, 256, kwt, 256, Mbase, Nbase, 8, smem, acc);
  int lane = threadIdx.x & 31, wave = threadIdx.x >> 5;
#pragma unroll
  for (int nt = 0; nt < 8; ++nt)
#pragma unroll
    for (int r = 0; r < 8; ++r) {
      long row = Mbase + wave * 16 + (lane >> 4) * 8 + r;
      long col = Nbase + nt * 16 + (lane & 15);
      kraw[row * 256 + col] = acc[nt][r];
    }
}

// =====================================================================
// Kernel 4: q projection (f32, exact) + per-head LN -> q_ln (2,N,128)
// One block per n; thread o computes q_raw[n][o] then group-LN per 128.
// =====================================================================
__global__ void __launch_bounds__(256) qproj_kernel(
    const float* __restrict__ x0f, const float* __restrict__ qw,
    const float* __restrict__ qnw, const float* __restrict__ qnb,
    float* __restrict__ qln) {
  __shared__ float xr[256];
  __shared__ float r1[256];
  __shared__ float r2[256];
  int n = blockIdx.x, t = threadIdx.x;
  xr[t] = x0f[n * 256 + t];
  __syncthreads();
  float dot = 0.f;
  const float* w = qw + (size_t)t * 256;
  for (int d = 0; d < 256; ++d) dot += xr[d] * w[d];
  r1[t] = dot; r2[t] = dot * dot;
  __syncthreads();
  for (int off = 64; off > 0; off >>= 1) {
    if ((t & 127) < off) { r1[t] += r1[t + off]; r2[t] += r2[t + off]; }
    __syncthreads();
  }
  int base = t & 128;
  float mu  = r1[base] * (1.0f / 128.0f);
  float var = r2[base] * (1.0f / 128.0f) - mu * mu;
  float rr  = rsqrtf(var + 1e-5f);
  int dq = t & 127, hd = t >> 7;
  qln[hd * NNq * 128 + n * 128 + dq] = (dot - mu) * rr * qnw[dq] + qnb[dq];
}

// =====================================================================
// Kernel 5: zero accumulators (graph-replay safe)
// =====================================================================
__global__ void zero_kernel(float* swacc, float* cnt) {
  int t = threadIdx.x;
  swacc[t] = 0.f; swacc[256 + t] = 0.f; cnt[t] = 0.f;
}

// =====================================================================
// Kernel 6: per-head LN of k rows + masked dot with q -> atomic partial sw
// One wave per (s,n) row of k_raw.
// =====================================================================
__global__ void __launch_bounds__(256) kln_sw_kernel(
    const float* __restrict__ kraw, const float* __restrict__ knw,
    const float* __restrict__ knb, const float* __restrict__ qln,
    const unsigned char* __restrict__ fullmask,
    float* __restrict__ swacc, float* __restrict__ cnt) {
  int wave = threadIdx.x >> 5, lane = threadIdx.x & 31;
  int row = blockIdx.x * 8 + wave;  // s*N + n
  int s = row >> 8, n = row & 255;
  const float* kr = kraw + (size_t)row * 256;
  float4 h0 = *(const float4*)(kr + lane * 4);
  float4 h1 = *(const float4*)(kr + 128 + lane * 4);
  float a0[4] = {h0.x, h0.y, h0.z, h0.w};
  float a1[4] = {h1.x, h1.y, h1.z, h1.w};
  float s0 = 0, q0 = 0, s1 = 0, q1 = 0;
#pragma unroll
  for (int u = 0; u < 4; ++u) {
    s0 += a0[u]; q0 += a0[u] * a0[u];
    s1 += a1[u]; q1 += a1[u] * a1[u];
  }
  s0 = wave_sum32(s0); q0 = wave_sum32(q0);
  s1 = wave_sum32(s1); q1 = wave_sum32(q1);
  float mu0 = s0 * (1.0f / 128.0f), var0 = q0 * (1.0f / 128.0f) - mu0 * mu0;
  float mu1 = s1 * (1.0f / 128.0f), var1 = q1 * (1.0f / 128.0f) - mu1 * mu1;
  float r0 = rsqrtf(var0 + 1e-5f), r1 = rsqrtf(var1 + 1e-5f);
  const float* q0p = qln + (size_t)n * 128;
  const float* q1p = qln + (size_t)(NNq + n) * 128;
  float d0 = 0.f, d1 = 0.f;
#pragma unroll
  for (int u = 0; u < 4; ++u) {
    int c = lane * 4 + u;
    float v0 = (a0[u] - mu0) * r0 * knw[c] + knb[c];
    float v1 = (a1[u] - mu1) * r1 * knw[c] + knb[c];
    d0 += v0 * q0p[c];
    d1 += v1 * q1p[c];
  }
  d0 = wave_sum32(d0);
  d1 = wave_sum32(d1);
  if (lane == 0) {
    float f = (float)fullmask[row];
    const float sc = 0.08838834764831845f;  // 128^-0.5
    atomicAdd(&swacc[s], d0 * sc * f);
    atomicAdd(&swacc[SS + s], d1 * sc * f);
    atomicAdd(&cnt[s], f);
  }
}

// =====================================================================
// Kernel 7: lambda + masked softmax over S -> seq_weights (output tail)
//           and sqrt(w+eps) scale table. Single block of 256.
// =====================================================================
__global__ void __launch_bounds__(256) softmax_kernel(
    const float* __restrict__ swacc, const float* __restrict__ cnt,
    const float* __restrict__ lq1, const float* __restrict__ lk1,
    const float* __restrict__ lq2, const float* __restrict__ lk2,
    const unsigned char* __restrict__ seqmask,
    float* __restrict__ out_seqw, float* __restrict__ sqscale) {
  __shared__ float red[256];
  int t = threadIdx.x;
  red[t] = (t < 128) ? lq1[t] * lk1[t] : 0.f;
  __syncthreads();
  for (int off = 128; off > 0; off >>= 1) {
    if (t < off) red[t] += red[t + off];
    __syncthreads();
  }
  float e1 = expf(red[0]);
  __syncthreads();
  red[t] = (t < 128) ? lq2[t] * lk2[t] : 0.f;
  __syncthreads();
  for (int off = 128; off > 0; off >>= 1) {
    if (t < off) red[t] += red[t + off];
    __syncthreads();
  }
  float lam = e1 - expf(red[0]) + LAMBDA_INIT;
  __syncthreads();
  float c = cnt[t] + EPSF;
  float v = (swacc[t] - lam * swacc[256 + t]) / c;
  if (!seqmask[t]) v = -1e9f;
  red[t] = v;
  __syncthreads();
  for (int off = 128; off > 0; off >>= 1) {
    if (t < off) red[t] = fmaxf(red[t], red[t + off]);
    __syncthreads();
  }
  float mx = red[0];
  __syncthreads();
  float e = expf(v - mx);
  red[t] = e;
  __syncthreads();
  for (int off = 128; off > 0; off >>= 1) {
    if (t < off) red[t] += red[t + off];
    __syncthreads();
  }
  float w = e / red[0];
  out_seqw[t] = w;
  sqscale[t] = sqrtf(w + EPSF);
}

// =====================================================================
// Kernel 8: a/bb projections, masked + sqrt-weight scaled, stored bf16:
//   at[(n*32+c)][s]   (A of outer GEMM, lda = S = 256)
//   bbm[s][(n*32+c)]  (B of outer GEMM, ldb = N*C = 8192)
// One wave per (s,n); lane = output channel c.
// =====================================================================
__global__ void __launch_bounds__(256) abproj_kernel(
    const __bf16* __restrict__ xbf, const float* __restrict__ wl,
    const float* __restrict__ wr, const unsigned char* __restrict__ fullmask,
    const float* __restrict__ sqscale,
    __bf16* __restrict__ at, __bf16* __restrict__ bbm) {
  __shared__ float xr[8][256];
  int wave = threadIdx.x >> 5, lane = threadIdx.x & 31;
  int row = blockIdx.x * 8 + wave;  // s*N + n
  int s = row >> 8, n = row & 255;
#pragma unroll
  for (int u = 0; u < 8; ++u)
    xr[wave][lane * 8 + u] = (float)xbf[(size_t)row * 256 + lane * 8 + u];
  __syncthreads();
  float da = 0.f, db = 0.f;
  const float* wlr = wl + lane * 256;
  const float* wrr = wr + lane * 256;
  for (int d = 0; d < 256; ++d) {
    float x = xr[wave][d];
    da += x * wlr[d];
    db += x * wrr[d];
  }
  float scl = (float)fullmask[row] * sqscale[s];
  at[((size_t)n * 32 + lane) * 256 + s] = (__bf16)(da * scl);
  bbm[((size_t)s * 256 + n) * 32 + lane] = (__bf16)(db * scl);
}

// =====================================================================
// Kernel 9: FUSED outer-product GEMM + wp GEMM + SiLU gating.
// Grid 64x64; block covers 4 i's x 4 j's:
//  Stage 1: 128x128x256 WMMA GEMM (double-buffered async tiles)
//           -> outer tile, staged in LDS as bf16 (reuses tile buffers).
//  Stage 2: 16(pairs) x 256 = outer_flat(16x1024) @ wpt(1024x256), WMMA.
//  Epilogue: bias + SiLU gate + pairwise mask -> out.
// LDS: [0,32K) stage-1 double buffers, later outer bf16; [32K,48K) pair f32.
// =====================================================================
__global__ void __launch_bounds__(256) outer_fused_kernel(
    const __bf16* __restrict__ at, const __bf16* __restrict__ bbm,
    const __bf16* __restrict__ wpt, const float* __restrict__ bp,
    const unsigned char* __restrict__ pmask, float* __restrict__ out) {
  __shared__ __align__(16) char smem[49152];
  __bf16* outerbf = (__bf16*)smem;           // 32KB, alias of tile buffers
  float* pairb = (float*)(smem + 32768);     // 16KB

  int t = threadIdx.x, lane = t & 31, wave = t >> 5;
  long Mbase = (long)blockIdx.x * 128, Nbase = (long)blockIdx.y * 128;

  // ---- Stage 1: outer tile ----
  v8f acc[8];
#pragma unroll
  for (int i = 0; i < 8; ++i) { v8f z = {}; acc[i] = z; }
  gemm128_db(at, 256, bbm, 8192, Mbase, Nbase, 8, smem, acc);
  // gemm ends with a barrier: tile buffers dead -> reuse as outer tile
#pragma unroll
  for (int nt = 0; nt < 8; ++nt)
#pragma unroll
    for (int r = 0; r < 8; ++r) {
      int rr = wave * 16 + (lane >> 4) * 8 + r;   // local row = i_loc*32 + c
      int cc = nt * 16 + (lane & 15);             // local col = j_loc*32 + e
      outerbf[rr * 128 + cc] = (__bf16)acc[nt][r];
    }
  __syncthreads();

  // ---- Stage 2: pair = outer_flat @ wp.T for 16 (i,j) pairs ----
  v8f p0 = {}, p1 = {};
  int pt0 = wave * 2;  // two 16-wide column tiles per wave -> 256 outputs
  int m = lane & 15, half = lane >> 4;
  for (int kk = 0; kk < 32; ++kk) {
    v16bf af;
#pragma unroll
    for (int e = 0; e < 16; ++e) {
      int kl = (e < 8 ? e : e + 8) + half * 8;
      int k2 = kk * 32 + kl;                       // flattened (c,e): c=k2>>5, e=k2&31
      int rrow = (m >> 2) * 32 + (k2 >> 5);
      int rcol = (m & 3) * 32 + (k2 & 31);
      af[e] = outerbf[rrow * 128 + rcol];
    }
    v16bf b0, b1;
#pragma unroll
    for (int e = 0; e < 16; ++e) {
      int kl = (e < 8 ? e : e + 8) + half * 8;
      const __bf16* wrow = wpt + (size_t)(kk * 32 + kl) * 256;
      b0[e] = wrow[pt0 * 16 + m];
      b1[e] = wrow[pt0 * 16 + 16 + m];
    }
    p0 = __builtin_amdgcn_wmma_f32_16x16x32_bf16(false, af, false, b0, (short)0, p0, false, false);
    p1 = __builtin_amdgcn_wmma_f32_16x16x32_bf16(false, af, false, b1, (short)0, p1, false, false);
  }
#pragma unroll
  for (int r = 0; r < 8; ++r) {
    int mrow = (lane >> 4) * 8 + r;  // pair index 0..15
    pairb[mrow * 256 + pt0 * 16 + (lane & 15)] = p0[r];
    pairb[mrow * 256 + (pt0 + 1) * 16 + (lane & 15)] = p1[r];
  }
  __syncthreads();

  // ---- Epilogue: bias, SiLU gate, pairwise mask ----
  int mi = t >> 4;            // pair index
  int pbase = (t & 15) * 8;   // 8 of 128 output channels
  long gi = blockIdx.x * 4 + (mi >> 2);
  long gj = blockIdx.y * 4 + (mi & 3);
  float pm = (float)pmask[gi * 256 + gj];
#pragma unroll
  for (int u = 0; u < 8; ++u) {
    int p = pbase + u;
    float xh = pairb[mi * 256 + p] + bp[p];
    float g  = pairb[mi * 256 + 128 + p] + bp[128 + p];
    float sig = 1.0f / (1.0f + expf(-g));
    out[(gi * 256 + gj) * 128 + p] = xh * g * sig * pm;
  }
}

// =====================================================================
// Launcher
// =====================================================================
extern "C" void kernel_launch(void* const* d_in, const int* in_sizes, int n_in,
                              void* d_out, int out_size, void* d_ws, size_t ws_size,
                              hipStream_t stream) {
  (void)in_sizes; (void)n_in; (void)out_size; (void)ws_size;
  const float* msa          = (const float*)d_in[0];
  const unsigned char* seqm = (const unsigned char*)d_in[1];
  const unsigned char* fullm= (const unsigned char*)d_in[2];
  const unsigned char* pmask= (const unsigned char*)d_in[3];
  const float* lnw = (const float*)d_in[4];
  const float* lnb = (const float*)d_in[5];
  const float* qw  = (const float*)d_in[6];
  const float* qnw = (const float*)d_in[7];
  const float* qnb = (const float*)d_in[8];
  const float* knw = (const float*)d_in[9];
  const float* knb = (const float*)d_in[10];
  const float* lq1 = (const float*)d_in[11];
  const float* lk1 = (const float*)d_in[12];
  const float* lq2 = (const float*)d_in[13];
  const float* lk2 = (const float*)d_in[14];
  const float* wl  = (const float*)d_in[15];
  const float* wr  = (const float*)d_in[16];
  const float* wp  = (const float*)d_in[17];
  const float* bp  = (const float*)d_in[18];
  const float* kw  = (const float*)d_in[19];  // k_w appended last in setup dict
  float* out = (float*)d_out;

  char* ws = (char*)d_ws;
  size_t off = 0;
  auto alloc = [&](size_t bytes) -> char* {
    char* p = ws + off;
    off += (bytes + 255) & ~(size_t)255;
    return p;
  };
  __bf16* xbf    = (__bf16*)alloc((size_t)SS * NNq * DDm * 2);   // 32 MB
  float*  x0f    = (float*)alloc((size_t)NNq * DDm * 4);
  __bf16* kwt    = (__bf16*)alloc(256 * 256 * 2);
  __bf16* wpt    = (__bf16*)alloc(1024 * 256 * 2);
  float*  kraw   = (float*)alloc((size_t)SS * NNq * 256 * 4);    // 64 MB
  float*  qln    = (float*)alloc(2 * NNq * 128 * 4);
  float*  swacc  = (float*)alloc(2 * SS * 4);
  float*  cnt    = (float*)alloc(SS * 4);
  float*  sqsc   = (float*)alloc(SS * 4);
  __bf16* at     = (__bf16*)alloc((size_t)NNq * CCc * SS * 2);   // 4 MB
  __bf16* bbm    = (__bf16*)alloc((size_t)SS * NNq * CCc * 2);   // 4 MB

  // 1) LayerNorm
  ln_kernel<<<SS * NNq / 8, 256, 0, stream>>>(msa, lnw, lnb, xbf, x0f);
  // 2) weight transposes -> bf16
  wprep_kernel<<<1024, 256, 0, stream>>>(kw, wp, kwt, wpt);
  // 3) zero partial-sum buffers (graph replay safe)
  zero_kernel<<<1, 256, 0, stream>>>(swacc, cnt);
  // 4) q projection + LN
  qproj_kernel<<<NNq, 256, 0, stream>>>(x0f, qw, qnw, qnb, qln);
  // 5) k projection GEMM (WMMA, async double-buffered tiles)
  kproj_gemm<<<dim3(SS * NNq / 128, 2), 256, 0, stream>>>(xbf, kwt, kraw);
  // 6) k LN + masked q.k partials
  kln_sw_kernel<<<SS * NNq / 8, 256, 0, stream>>>(kraw, knw, knb, qln, fullm, swacc, cnt);
  // 7) lambda + softmax -> seq_weights (second output) + sqrt scale
  softmax_kernel<<<1, 256, 0, stream>>>(swacc, cnt, lq1, lk1, lq2, lk2, seqm,
                                        out + (size_t)NNq * NNq * DPp, sqsc);
  // 8) a / bb projections (scaled, masked, bf16, GEMM-friendly layouts)
  abproj_kernel<<<SS * NNq / 8, 256, 0, stream>>>(xbf, wl, wr, fullm, sqsc, at, bbm);
  // 9) fused outer-product GEMM + wp GEMM + SiLU (WMMA x2 stages)
  outer_fused_kernel<<<dim3(NNq / 4, NNq / 4), 256, 0, stream>>>(at, bbm, wpt, bp, pmask, out);
}
